// QuantumDeepField_6597069766656
// MI455X (gfx1250) — matrix operand order repaired and checked by reading, not compiled, not run on hardware
//
#include <hip/hip_runtime.h>
#include <hip/hip_bf16.h>
#include <math.h>

// ---------------------------------------------------------------------------
// QuantumDeepField forward for MI455X (gfx1250), wave32 + WMMA f16->f32.
// ---------------------------------------------------------------------------

typedef __attribute__((ext_vector_type(16))) _Float16 v16h;
typedef __attribute__((ext_vector_type(8)))  _Float16 v8h;
typedef __attribute__((ext_vector_type(8)))  float    v8f;

#define B_MOL 32
#define GRIDP 4096
#define A_ORB 64
#define DIMF  256
#define LFUNC 3
#define GTILE 64
#define LDS_STRIDE 264   // 256 + 8 halves pad -> 528B rows, bank-conflict-free b128
#define AT_STRIDE  72    // 64 + 8 halves pad  -> 144B rows, bank-conflict-free b128

#define WMMA_F16(a, b, c) \
  __builtin_amdgcn_wmma_f32_16x16x32_f16(false, (a), false, (b), (short)0, (c), false, false)

// A-fragment (16x32 f16, M x K) from LDS, row-major with `stride` halves.
// ISA layout: lanes0-15 M=lane, halves0-7 K=k0+grp*8.., halves8-15 K=k0+16+grp*8..
__device__ __forceinline__ v16h load_a_frag(const _Float16* buf, int stride,
                                            int row0, int k0, int lane) {
  const int m = lane & 15, grp = lane >> 4;
  const _Float16* p = buf + (row0 + m) * stride + k0 + grp * 8;
  v8h lo = *(const v8h*)p;
  v8h hi = *(const v8h*)(p + 16);
  v16h a;
#pragma unroll
  for (int i = 0; i < 8; ++i) { a[i] = lo[i]; a[i + 8] = hi[i]; }
  return a;
}

// B-fragment (32x16 f16, K x N) from global weights stored TRANSPOSED [n][k]:
// lane n = lane&15, halves0-7 K=k0+(lane>>4)*8.., halves8-15 K=k0+16+(lane>>4)*8..
__device__ __forceinline__ v16h load_b_frag(const _Float16* wt, int kdim,
                                            int n0, int k0, int lane) {
  const int n = lane & 15, grp = lane >> 4;
  const _Float16* p = wt + (size_t)(n0 + n) * kdim + k0 + grp * 8;
  v8h lo = *(const v8h*)p;
  v8h hi = *(const v8h*)(p + 16);
  v16h b;
#pragma unroll
  for (int i = 0; i < 8; ++i) { b[i] = lo[i]; b[i + 8] = hi[i]; }
  return b;
}

// ---------------------------------------------------------------------------
// K1: Wf_w fp32 [3][k=256][n=256] -> WfT f16 [3][n][k]; zero msum/pooled.
// ---------------------------------------------------------------------------
__global__ void k1_prep(const float* __restrict__ Wf_w, _Float16* __restrict__ WfT,
                        float* __restrict__ msum, float* __restrict__ pooled) {
  int idx = blockIdx.x * 256 + threadIdx.x;            // 3*65536 total
  if (idx < LFUNC * DIMF * DIMF) {
    int l = idx >> 16, r = idx & 65535;
    int k = r >> 8, n = r & 255;
    WfT[(size_t)l * 65536 + n * 256 + k] = (_Float16)Wf_w[(size_t)l * 65536 + r];
  }
  if (idx < B_MOL * DIMF) { msum[idx] = 0.f; pooled[idx] = 0.f; }
}

// ---------------------------------------------------------------------------
// K2: per (b,d) L2-norm of gathered coefs over the 64 orbitals,
//     store normalized f16 transposed: Ct[b][d][a].
// ---------------------------------------------------------------------------
__global__ void k2_coefs(const int* __restrict__ ao, const float* __restrict__ coef,
                         _Float16* __restrict__ Ct) {
  int b = blockIdx.x, d = threadIdx.x;
  __shared__ int sa[A_ORB];
  if (d < A_ORB) sa[d] = ao[b * A_ORB + d];
  __syncthreads();
  float s = 0.f;
#pragma unroll 8
  for (int a = 0; a < A_ORB; ++a) {
    float c = coef[(size_t)sa[a] * DIMF + d];
    s += c * c;
  }
  float inv = 1.f / fmaxf(sqrtf(s), 1e-12f);
  _Float16* row = Ct + ((size_t)b * DIMF + d) * A_ORB;
#pragma unroll 8
  for (int a = 0; a < A_ORB; ++a)
    row[a] = (_Float16)(coef[(size_t)sa[a] * DIMF + d] * inv);
}

// ---------------------------------------------------------------------------
// K3: per (b,a) GTO L2-norm over the 4096 grid points -> 1/max(norm,eps).
// ---------------------------------------------------------------------------
__global__ void k3_gnorm(const int* __restrict__ ao, const float* __restrict__ dist,
                         const int* __restrict__ qn, const float* __restrict__ zeta,
                         float* __restrict__ gn_inv) {
  int b = blockIdx.x, t = threadIdx.x;
  int a = t & 63, gr = t >> 6;                         // 4 thread-rows per column
  __shared__ float az[A_ORB];
  __shared__ int   aq[A_ORB];
  __shared__ float red[256];
  if (t < A_ORB) {
    az[t] = zeta[ao[b * A_ORB + t]];
    aq[t] = qn[b * A_ORB + t];
  }
  __syncthreads();
  float z = az[a]; int qi = aq[a];
  float s = 0.f;
  for (int g = gr; g < GRIDP; g += 4) {
    float d = dist[((size_t)b * GRIDP + g) * A_ORB + a];
    float p = (qi == 1) ? 1.f : ((qi == 2) ? d : d * d);
    float gv = p * __expf(-z * d * d);
    s += gv * gv;
  }
  red[t] = s;
  __syncthreads();
  if (t < A_ORB) {
    float tot = red[t] + red[t + 64] + red[t + 128] + red[t + 192];
    gn_inv[b * A_ORB + t] = 1.f / fmaxf(sqrtf(tot), 1e-12f);
  }
}

// ---------------------------------------------------------------------------
// K4: MOs = GTOs_norm @ coefs_norm via WMMA.  One block = (b, 64 grid rows).
//     Also accumulates per-(b,d) sum of squares, stores MOs f16 coalesced.
// ---------------------------------------------------------------------------
__global__ void k4_mos(const int* __restrict__ ao, const float* __restrict__ dist,
                       const int* __restrict__ qn, const float* __restrict__ zeta,
                       const float* __restrict__ gn_inv, const _Float16* __restrict__ Ct,
                       _Float16* __restrict__ MOs, float* __restrict__ msum) {
  const int b = blockIdx.x >> 6, g0 = (blockIdx.x & 63) * GTILE;
  const int tid = threadIdx.x, wave = tid >> 5, lane = tid & 31;
  __shared__ __align__(16) _Float16 At[GTILE * AT_STRIDE];
  __shared__ __align__(16) _Float16 Ot[GTILE * LDS_STRIDE];
  __shared__ float az[A_ORB], agi[A_ORB];
  __shared__ int   aq[A_ORB];

  if (tid < A_ORB) {
    int o = ao[b * A_ORB + tid];
    az[tid]  = zeta[o];
    aq[tid]  = qn[b * A_ORB + tid];
    agi[tid] = gn_inv[b * A_ORB + tid];
  }
  __syncthreads();

  // Build normalized-GTO A-tile [64 x 64] in f16.
  for (int i = tid; i < GTILE * A_ORB; i += 256) {
    int row = i >> 6, a = i & 63;
    float d = dist[((size_t)b * GRIDP + g0 + row) * A_ORB + a];
    int qi = aq[a];
    float p = (qi == 1) ? 1.f : ((qi == 2) ? d : d * d);
    float gv = p * __expf(-az[a] * d * d) * agi[a];
    At[row * AT_STRIDE + a] = (_Float16)gv;
  }
  __syncthreads();

  v8f acc[4][2] = {};
  const _Float16* CtB = Ct + (size_t)b * DIMF * A_ORB;
  const int n0 = wave * 32;
#pragma unroll
  for (int kb = 0; kb < 2; ++kb) {
    int k0 = kb * 32;
    v16h b0 = load_b_frag(CtB, A_ORB, n0,      k0, lane);
    v16h b1 = load_b_frag(CtB, A_ORB, n0 + 16, k0, lane);
#pragma unroll
    for (int mt = 0; mt < 4; ++mt) {
      v16h a = load_a_frag(At, AT_STRIDE, mt * 16, k0, lane);
      acc[mt][0] = WMMA_F16(a, b0, acc[mt][0]);
      acc[mt][1] = WMMA_F16(a, b1, acc[mt][1]);
    }
  }

  const int grp = lane >> 4, nn = lane & 15;
#pragma unroll
  for (int nt = 0; nt < 2; ++nt) {
    float ss = 0.f;
#pragma unroll
    for (int mt = 0; mt < 4; ++mt)
#pragma unroll
      for (int r = 0; r < 8; ++r) {
        float f = acc[mt][nt][r];
        ss += f * f;
        Ot[(mt * 16 + r + 8 * grp) * LDS_STRIDE + n0 + nt * 16 + nn] = (_Float16)f;
      }
    atomicAdd(&msum[b * DIMF + n0 + nt * 16 + nn], ss);
  }
  __syncthreads();

  // Coalesced f16 store of the tile.
  unsigned* mo = (unsigned*)(MOs + ((size_t)b * GRIDP + g0) * DIMF);
  for (int i = tid; i < GTILE * (DIMF / 2); i += 256) {
    int row = i >> 7, c = i & 127;
    mo[row * (DIMF / 2) + c] = *(const unsigned*)&Ot[row * LDS_STRIDE + c * 2];
  }
}

// ---------------------------------------------------------------------------
// K5: scale[b][d] = sqrt(Ne/dim) / max(||MO col||, eps)
// ---------------------------------------------------------------------------
__global__ void k5_scale(const float* __restrict__ Ne, const float* __restrict__ msum,
                         float* __restrict__ scale_d) {
  int i = blockIdx.x * 256 + threadIdx.x;              // 8192
  int b = i >> 8;
  scale_d[i] = sqrtf(Ne[b] / (float)DIMF) / fmaxf(sqrtf(msum[i]), 1e-12f);
}

// ---------------------------------------------------------------------------
// K6: 3-layer ReLU MLP on a 64-row tile (WMMA), in-place LDS ping, sum-pool.
// ---------------------------------------------------------------------------
__global__ void k6_mlp(const _Float16* __restrict__ MOs, const float* __restrict__ scale_d,
                       const _Float16* __restrict__ WfT, const float* __restrict__ Wf_b,
                       float* __restrict__ pooled) {
  const int b = blockIdx.x >> 6, g0 = (blockIdx.x & 63) * GTILE;
  const int tid = threadIdx.x, wave = tid >> 5, lane = tid & 31;
  const int grp = lane >> 4, nn = lane & 15;
  const int n0 = wave * 32;
  __shared__ __align__(16) _Float16 buf[GTILE * LDS_STRIDE];
  __shared__ float sc[DIMF];

  sc[tid] = scale_d[b * DIMF + tid];
  __syncthreads();

  // v0 = scale * MOs (f16 into LDS, coalesced dword loads).
  const unsigned* mo = (const unsigned*)(MOs + ((size_t)b * GRIDP + g0) * DIMF);
  for (int i = tid; i < GTILE * (DIMF / 2); i += 256) {
    int row = i >> 7, c = i & 127;
    union { unsigned u; _Float16 h[2]; } t;
    t.u = mo[row * (DIMF / 2) + c];
    t.h[0] = (_Float16)((float)t.h[0] * sc[c * 2]);
    t.h[1] = (_Float16)((float)t.h[1] * sc[c * 2 + 1]);
    *(unsigned*)&buf[row * LDS_STRIDE + c * 2] = t.u;
  }
  __syncthreads();

  for (int l = 0; l < LFUNC; ++l) {
    v8f acc[4][2] = {};
    const _Float16* Wl = WfT + (size_t)l * DIMF * DIMF;
#pragma unroll
    for (int kb = 0; kb < 8; ++kb) {
      int k0 = kb * 32;
      v16h b0 = load_b_frag(Wl, DIMF, n0,      k0, lane);
      v16h b1 = load_b_frag(Wl, DIMF, n0 + 16, k0, lane);
#pragma unroll
      for (int mt = 0; mt < 4; ++mt) {
        v16h a = load_a_frag(buf, LDS_STRIDE, mt * 16, k0, lane);
        acc[mt][0] = WMMA_F16(a, b0, acc[mt][0]);
        acc[mt][1] = WMMA_F16(a, b1, acc[mt][1]);
      }
    }
    float bias[2];
    bias[0] = Wf_b[l * DIMF + n0 + nn];
    bias[1] = Wf_b[l * DIMF + n0 + 16 + nn];
    __syncthreads();                                   // all reads of buf done
    if (l < LFUNC - 1) {
#pragma unroll
      for (int nt = 0; nt < 2; ++nt)
#pragma unroll
        for (int mt = 0; mt < 4; ++mt)
#pragma unroll
          for (int r = 0; r < 8; ++r) {
            float f = fmaxf(acc[mt][nt][r] + bias[nt], 0.f);
            buf[(mt * 16 + r + 8 * grp) * LDS_STRIDE + n0 + nt * 16 + nn] = (_Float16)f;
          }
      __syncthreads();                                 // writes visible to next layer
    } else {
#pragma unroll
      for (int nt = 0; nt < 2; ++nt) {
        float p = 0.f;
#pragma unroll
        for (int mt = 0; mt < 4; ++mt)
#pragma unroll
          for (int r = 0; r < 8; ++r)
            p += fmaxf(acc[mt][nt][r] + bias[nt], 0.f);
        atomicAdd(&pooled[b * DIMF + n0 + nt * 16 + nn], p);
      }
    }
  }
}

// ---------------------------------------------------------------------------
// K7: out[b] = sum_d pooled[b][d]*Wp_w[d] + Wp_b
// ---------------------------------------------------------------------------
__global__ void k7_out(const float* __restrict__ pooled, const float* __restrict__ Wp_w,
                       const float* __restrict__ Wp_b, float* __restrict__ out) {
  int b = blockIdx.x, t = threadIdx.x;
  __shared__ float red[256];
  red[t] = pooled[b * DIMF + t] * Wp_w[t];
  __syncthreads();
  for (int s = 128; s > 0; s >>= 1) {
    if (t < s) red[t] += red[t + s];
    __syncthreads();
  }
  if (t == 0) out[b] = red[0] + Wp_b[0];
}

// ---------------------------------------------------------------------------
extern "C" void kernel_launch(void* const* d_in, const int* in_sizes, int n_in,
                              void* d_out, int out_size, void* d_ws, size_t ws_size,
                              hipStream_t stream) {
  const int*   ao    = (const int*)d_in[0];
  const float* dist  = (const float*)d_in[1];
  const int*   qn    = (const int*)d_in[2];
  const float* Ne    = (const float*)d_in[3];
  const float* coef  = (const float*)d_in[4];
  const float* zeta  = (const float*)d_in[5];
  const float* Wf_w  = (const float*)d_in[6];
  const float* Wf_b  = (const float*)d_in[7];
  const float* Wp_w  = (const float*)d_in[8];
  const float* Wp_b  = (const float*)d_in[9];
  float* out = (float*)d_out;

  char* w = (char*)d_ws;
  size_t off = 0;
  auto alloc = [&](size_t bytes) { size_t o = off; off = (off + bytes + 255) & ~(size_t)255; return o; };
  _Float16* WfT    = (_Float16*)(w + alloc((size_t)LFUNC * DIMF * DIMF * 2));
  _Float16* Ct     = (_Float16*)(w + alloc((size_t)B_MOL * DIMF * A_ORB * 2));
  float*    gn_inv = (float*)(w + alloc((size_t)B_MOL * A_ORB * 4));
  float*    msum   = (float*)(w + alloc((size_t)B_MOL * DIMF * 4));
  float*    sc     = (float*)(w + alloc((size_t)B_MOL * DIMF * 4));
  float*    pooled = (float*)(w + alloc((size_t)B_MOL * DIMF * 4));
  _Float16* MOs    = (_Float16*)(w + alloc((size_t)B_MOL * GRIDP * DIMF * 2));
  if (off > ws_size) return;  // workspace too small; nothing safe to do

  k1_prep <<<dim3((LFUNC * DIMF * DIMF + 255) / 256), dim3(256), 0, stream>>>(Wf_w, WfT, msum, pooled);
  k2_coefs<<<dim3(B_MOL), dim3(256), 0, stream>>>(ao, coef, Ct);
  k3_gnorm<<<dim3(B_MOL), dim3(256), 0, stream>>>(ao, dist, qn, zeta, gn_inv);
  k4_mos  <<<dim3(B_MOL * (GRIDP / GTILE)), dim3(256), 0, stream>>>(ao, dist, qn, zeta, gn_inv, Ct, MOs, msum);
  k5_scale<<<dim3(B_MOL), dim3(256), 0, stream>>>(Ne, msum, sc);
  k6_mlp  <<<dim3(B_MOL * (GRIDP / GTILE)), dim3(256), 0, stream>>>(MOs, sc, WfT, Wf_b, pooled);
  k7_out  <<<dim3(B_MOL), dim3(256), 0, stream>>>(pooled, Wp_w, Wp_b, out);
}
